// Quantizer_85358180040916
// MI455X (gfx1250) — matrix operand order repaired and checked by reading, not compiled, not run on hardware
//
#include <hip/hip_runtime.h>

typedef __bf16 bf16x16 __attribute__((ext_vector_type(16)));
typedef float v8f __attribute__((ext_vector_type(8)));

constexpr int   D     = 256;
constexpr int   V     = 8192;
constexpr int   SP    = 4096;        // 16^3
constexpr int   NPTS  = 32768;       // 8 * 16^3
constexpr float GAMMA_F = 0.99f;
constexpr float ONEMG   = 1.0f - 0.99f;
constexpr float EPS_F   = 1e-7f;

// d_out layout (floats), in reference return order
constexpr size_t OUT_QUANT = 0;                       // 8*256*4096 = 8388608
constexpr size_t OUT_IDX   = 8388608;                 // 32768
constexpr size_t OUT_DIFF  = OUT_IDX + 32768;         // 1
constexpr size_t OUT_N     = OUT_DIFF + 1;            // 8192
constexpr size_t OUT_ZAVG  = OUT_N + 8192;            // 8192*256
constexpr size_t OUT_W     = OUT_ZAVG + (size_t)V*D;  // 8192*256

constexpr int LDS_STRIDE = 264;      // bf16 units; 264*2B=528B => conflict-free b128
constexpr int NTILES     = V / 64;   // 128

__device__ __forceinline__ unsigned short f2bf(float f) {
    union { float f; unsigned u; } v; v.f = f;
    unsigned u = v.u;
    unsigned r = u + 0x7FFFu + ((u >> 16) & 1u);   // round-to-nearest-even
    return (unsigned short)(r >> 16);
}

// ---- CDNA5 async global->LDS copy (no VGPR data path, ASYNCcnt) -----------
// VDST = LDS byte address (low 32 bits of flat pointer, aperture truncation),
// VADDR = 64-bit global address, SADDR = off.  Tracked by ASYNCcnt.
__device__ __forceinline__ void async_copy_b128(const unsigned short* g,
                                                unsigned short* l) {
    unsigned           lo = (unsigned)(uintptr_t)l;            // LDS byte offset
    unsigned long long ga = (unsigned long long)(uintptr_t)g;  // global address
    asm volatile("global_load_async_to_lds_b128 %0, %1, off"
                 :: "v"(lo), "v"(ga) : "memory");
}

__device__ __forceinline__ void wait_async0() {
#if __has_builtin(__builtin_amdgcn_s_wait_asynccnt)
    __builtin_amdgcn_s_wait_asynccnt(0);
#else
    asm volatile("s_wait_asynccnt 0x0" ::: "memory");
#endif
}

// ---- pre-pass: codebook -> bf16 + ||w||^2 --------------------------------
__global__ void vq_prep_w(const float* __restrict__ w,
                          unsigned short* __restrict__ wbf,
                          float* __restrict__ wnorm) {
    int v = blockIdx.x, d = threadIdx.x;
    float val = w[(size_t)v * D + d];
    wbf[(size_t)v * D + d] = f2bf(val);
    __shared__ float red[256];
    red[d] = val * val; __syncthreads();
    for (int s = 128; s > 0; s >>= 1) {
        if (d < s) red[d] += red[d + s];
        __syncthreads();
    }
    if (d == 0) wnorm[v] = red[0];
}

// ---- pre-pass: x (b,D,sp) -> flat (i,D) bf16, channels-last ---------------
__global__ void vq_prep_x(const float* __restrict__ x,
                          unsigned short* __restrict__ flatbf) {
    int i = blockIdx.x, d = threadIdx.x;
    int b = i >> 12, sp = i & (SP - 1);
    float v = x[((size_t)(b * D + d)) * SP + sp];
    flatbf[(size_t)i * D + d] = f2bf(v);
}

// ---- init EMA outputs with gamma * state, zero scalars --------------------
__global__ void vq_init(const float* __restrict__ Nin,
                        const float* __restrict__ zin,
                        float* __restrict__ out,
                        float* __restrict__ nsum) {
    size_t j = (size_t)blockIdx.x * 256 + threadIdx.x;   // covers V*D
    out[OUT_ZAVG + j] = GAMMA_F * zin[j];
    if (j < V)  out[OUT_N + j] = GAMMA_F * Nin[j];
    if (j == 0) { out[OUT_DIFF] = 0.0f; *nsum = 0.0f; }
}

// ---- main WMMA argmin: maximize dot(x,w) - 0.5*||w||^2 --------------------
// Double-buffered LDS B tiles filled by async global->LDS DMA, 1 barrier/iter.
__global__ void __launch_bounds__(256)
vq_argmin(const unsigned short* __restrict__ flatbf,
          const unsigned short* __restrict__ wbf,
          const float* __restrict__ wnorm,
          int* __restrict__ widx,
          float* __restrict__ idxf) {
    __shared__ unsigned short ldsB[2][64 * LDS_STRIDE];   // 2 x 33.8 KB
    const int tid  = threadIdx.x;
    const int lane = tid & 31;
    const int wave = tid >> 5;
    const int l15  = lane & 15;
    const int half = lane >> 4;
    const int m0   = blockIdx.x * 128 + wave * 16;

    union AB { uint4 q[2]; bf16x16 v; };

    // A tile: 16 rows x 256 K held in registers (64 VGPRs), ISA 16-bit layout
    AB afrag[8];
    {
        const unsigned short* ap = flatbf + (size_t)(m0 + l15) * D + half * 8;
#pragma unroll
        for (int ks = 0; ks < 8; ++ks) {
            afrag[ks].q[0] = *(const uint4*)(ap + ks * 32);        // K: base..+7
            afrag[ks].q[1] = *(const uint4*)(ap + ks * 32 + 16);   // K: base+16..+23
        }
    }

    // async-DMA one 64-codeword tile into an LDS buffer (no VGPR staging)
    auto async_tile = [&](int t, int buf) {
        const int n0 = t * 64;
#pragma unroll
        for (int j = 0; j < 8; ++j) {
            int lin = tid + j * 256;             // uint4 units within tile
            int r = lin >> 5, c16 = lin & 31;
            async_copy_b128(wbf + (size_t)(n0 + r) * D + c16 * 8,
                            &ldsB[buf][r * LDS_STRIDE + c16 * 8]);
        }
    };

    float wn_cur[4], wn_next[4];
    float bestv[8]; int besti[8];
#pragma unroll
    for (int r = 0; r < 8; ++r) { bestv[r] = -3.4e38f; besti[r] = 0; }

    // prologue: tile 0 -> LDS buffer 0
    async_tile(0, 0);
#pragma unroll
    for (int nc = 0; nc < 4; ++nc) wn_cur[nc] = wnorm[nc * 16 + l15];
    wait_async0();
    __syncthreads();

    for (int t = 0; t < NTILES; ++t) {
        const int cur = t & 1;
        const int n0  = t * 64;

        // kick off next tile's DMA into the alternate buffer (safe: last read
        // at iter t-1, protected by that iteration's barrier)
        if (t + 1 < NTILES) {
            async_tile(t + 1, cur ^ 1);
#pragma unroll
            for (int nc = 0; nc < 4; ++nc)
                wn_next[nc] = wnorm[(t + 1) * 64 + nc * 16 + l15];
        }

#pragma unroll
        for (int nc = 0; nc < 4; ++nc) {
            const int cw  = nc * 16 + l15;
            const int col = n0 + cw;
            const float wn = wn_cur[nc];
            v8f c;
#pragma unroll
            for (int r = 0; r < 8; ++r) c[r] = -0.5f * wn;   // fold ||w||^2

            const unsigned short* bp = &ldsB[cur][cw * LDS_STRIDE + half * 8];
#pragma unroll
            for (int ks = 0; ks < 8; ++ks) {
                AB bfrag;
                bfrag.q[0] = *(const uint4*)(bp + ks * 32);       // K: base..+7
                bfrag.q[1] = *(const uint4*)(bp + ks * 32 + 16);  // K: base+16..+23
                c = __builtin_amdgcn_wmma_f32_16x16x32_bf16(
                        false, afrag[ks].v, false, bfrag.v,
                        (short)0, c, false, false);
            }
#pragma unroll
            for (int r = 0; r < 8; ++r) {
                float s = c[r];
                if (s > bestv[r]) { bestv[r] = s; besti[r] = col; }
            }
        }

        if (t + 1 < NTILES) {
            wait_async0();            // my DMA into alt buffer done
            __syncthreads();          // everyone's DMA + reads of cur done
#pragma unroll
            for (int nc = 0; nc < 4; ++nc) wn_cur[nc] = wn_next[nc];
        }
    }

    // reduce across the 16 lanes of each half (C: lanes0-15 rows r, 16-31 rows r+8)
#pragma unroll
    for (int r = 0; r < 8; ++r) {
#pragma unroll
        for (int m = 8; m >= 1; m >>= 1) {
            float ov = __shfl_xor(bestv[r], m, 32);
            int   oi = __shfl_xor(besti[r], m, 32);
            if (ov > bestv[r] || (ov == bestv[r] && oi < besti[r])) {
                bestv[r] = ov; besti[r] = oi;
            }
        }
    }
    if (l15 == 0) {
#pragma unroll
        for (int r = 0; r < 8; ++r) {
            int row = m0 + half * 8 + r;
            widx[row] = besti[r];
            idxf[row] = (float)besti[r];
        }
    }
}

// ---- EMA scatter: N += (1-g), z_avg += (1-g)*x ----------------------------
__global__ void vq_scatter(const float* __restrict__ x,
                           const int* __restrict__ widx,
                           float* __restrict__ nout,
                           float* __restrict__ zout) {
    int i = blockIdx.x, d = threadIdx.x;
    int idx = widx[i];
    int b = i >> 12, sp = i & (SP - 1);
    float xv = x[((size_t)(b * D + d)) * SP + sp];
    atomicAdd(&zout[(size_t)idx * D + d], ONEMG * xv);
    if (d == 0) atomicAdd(&nout[idx], ONEMG);
}

// ---- quant gather (coalesced) + MSE ---------------------------------------
__global__ void vq_quant(const float* __restrict__ x,
                         const float* __restrict__ w,
                         const int* __restrict__ widx,
                         float* __restrict__ qout,
                         float* __restrict__ diffp) {
    int bd = blockIdx.x;            // b*256 + d
    int b = bd >> 8, d = bd & 255;
    const float* xp = x + (size_t)bd * SP;
    float* qp = qout + (size_t)bd * SP;
    float acc = 0.0f;
    for (int k = 0; k < SP; k += 256) {
        int sp = k + threadIdx.x;
        int idx = widx[b * SP + sp];
        float q  = w[(size_t)idx * D + d];
        float xv = xp[sp];
        qp[sp] = q;                 // quant_st == quant numerically
        float df = xv - q; acc += df * df;
    }
    __shared__ float red[256];
    red[threadIdx.x] = acc; __syncthreads();
    for (int s = 128; s > 0; s >>= 1) {
        if (threadIdx.x < s) red[threadIdx.x] += red[threadIdx.x + s];
        __syncthreads();
    }
    if (threadIdx.x == 0) atomicAdd(diffp, red[0] * (1.0f / 8388608.0f));
}

// ---- n = sum(new_N) -------------------------------------------------------
__global__ void vq_nsum(const float* __restrict__ nout, float* __restrict__ nsum) {
    int v = blockIdx.x * 256 + threadIdx.x;
    __shared__ float red[256];
    red[threadIdx.x] = nout[v]; __syncthreads();
    for (int s = 128; s > 0; s >>= 1) {
        if (threadIdx.x < s) red[threadIdx.x] += red[threadIdx.x + s];
        __syncthreads();
    }
    if (threadIdx.x == 0) atomicAdd(nsum, red[0]);
}

// ---- new_weight = new_z_avg / ((N+eps)/(n+V*eps)*n) ----------------------
__global__ void vq_wupd(const float* __restrict__ nout,
                        const float* __restrict__ zout,
                        const float* __restrict__ nsum,
                        float* __restrict__ wout) {
    int v = blockIdx.x, d = threadIdx.x;
    float n = *nsum;
    float wgt = (nout[v] + EPS_F) / (n + (float)V * EPS_F) * n;
    wout[(size_t)v * D + d] = zout[(size_t)v * D + d] / wgt;
}

extern "C" void kernel_launch(void* const* d_in, const int* in_sizes, int n_in,
                              void* d_out, int out_size, void* d_ws, size_t ws_size,
                              hipStream_t stream) {
    (void)in_sizes; (void)n_in; (void)out_size; (void)ws_size;
    const float* x      = (const float*)d_in[0];
    const float* weight = (const float*)d_in[1];
    const float* Nin    = (const float*)d_in[2];
    const float* zin    = (const float*)d_in[3];
    float* out = (float*)d_out;

    char* ws = (char*)d_ws;
    unsigned short* wbf    = (unsigned short*)(ws);
    unsigned short* flatbf = (unsigned short*)(ws + (size_t)V * D * 2);
    float* wnorm = (float*)(ws + (size_t)V * D * 2 + (size_t)NPTS * D * 2);
    int*   widx  = (int*)((char*)wnorm + (size_t)V * 4);
    float* nsum  = (float*)((char*)widx + (size_t)NPTS * 4);

    vq_prep_w <<<V,          256, 0, stream>>>(weight, wbf, wnorm);
    vq_prep_x <<<NPTS,       256, 0, stream>>>(x, flatbf);
    vq_init   <<<(V*D)/256,  256, 0, stream>>>(Nin, zin, out, nsum);
    vq_argmin <<<NPTS/128,   256, 0, stream>>>(flatbf, wbf, wnorm, widx, out + OUT_IDX);
    vq_scatter<<<NPTS,       256, 0, stream>>>(x, widx, out + OUT_N, out + OUT_ZAVG);
    vq_quant  <<<2048,       256, 0, stream>>>(x, weight, widx, out + OUT_QUANT, out + OUT_DIFF);
    vq_nsum   <<<V/256,      256, 0, stream>>>(out + OUT_N, nsum);
    vq_wupd   <<<V,          256, 0, stream>>>(out + OUT_N, out + OUT_ZAVG, nsum, out + OUT_W);
}
